// SmallHGRNLM_17703855194217
// MI455X (gfx1250) — matrix operand order repaired and checked by reading, not compiled, not run on hardware
//
#include <hip/hip_runtime.h>
#include <cstdint>
#include <cstddef>

// ---------------------------------------------------------------------------
// SmallHGRNLM on MI455X (gfx1250): bf16 WMMA GEMMs + fused gate/scan kernel.
//   V=50000, H=256, L=2, B=4, T=1024, M = B*T = 4096
// Dominant cost: logits = X @ Wp  (104.9 GFLOP, ~875 MB traffic) -> bf16
// operands + f32 accumulation via v_wmma_f32_16x16x32_bf16; Wp (bf16, 25.6MB)
// stays L2-resident, logits stream out with non-temporal stores.
// ---------------------------------------------------------------------------

typedef __attribute__((ext_vector_type(16))) __bf16 v16bf;
typedef __attribute__((ext_vector_type(8)))  __bf16 v8bf;
typedef __attribute__((ext_vector_type(8)))  float  v8f;

// ---------------------------------------------------------------------------
// Weight convert + transpose:  W[K,N] fp32  ->  Wt[Npad,K] bf16 (zero-padded)
// ---------------------------------------------------------------------------
__global__ void convT_kernel(const float* __restrict__ W, __bf16* __restrict__ Wt,
                             int K, int N, int Npad) {
  int idx = blockIdx.x * blockDim.x + threadIdx.x;   // over Npad*K
  int n = idx / K;
  int k = idx - n * K;
  if (n >= Npad) return;
  float v = (n < N) ? W[(size_t)k * N + n] : 0.0f;
  Wt[idx] = (__bf16)v;
}

// ---------------------------------------------------------------------------
// Embedding gather: x = emb[ids], fp32 + bf16 shadow
// ---------------------------------------------------------------------------
__global__ void embed_kernel(const int* __restrict__ ids, const float* __restrict__ emb,
                             float* __restrict__ xf, __bf16* __restrict__ xbf,
                             int M, int H) {
  int idx = blockIdx.x * blockDim.x + threadIdx.x;   // over M*H
  int tok = idx / H;
  int h   = idx - tok * H;
  float v = emb[(size_t)ids[tok] * H + h];
  xf[idx]  = v;
  xbf[idx] = (__bf16)v;
}

// ---------------------------------------------------------------------------
// Fused gate activation + bounded-forget recurrence.
// One thread per (b,h) channel scans T steps:
//   i = silu(zi); f = lower + (1-lower)*sigmoid(zf); g = sigmoid(zg)
//   h_t = f*h_{t-1} + (1-f)*i ;  y_t = h_t * g  (bf16, feeds Wo GEMM)
// Loop-carried dep is 2 VALU ops; unroll so loads of future timesteps are in
// flight while the recurrence chain retires.
// ---------------------------------------------------------------------------
__global__ void scan_kernel(const float* __restrict__ zi, const float* __restrict__ zf,
                            const float* __restrict__ zg, __bf16* __restrict__ ybf,
                            float* __restrict__ hlast,
                            int B, int T, int H, float lower) {
  int tid = blockIdx.x * blockDim.x + threadIdx.x;   // over B*H
  int b = tid / H;
  int h = tid - b * H;
  const float upper = 1.0f - lower;
  float hs = 0.0f;
  size_t base = (size_t)b * T * H + h;
#pragma unroll 4
  for (int t = 0; t < T; ++t) {
    size_t idx = base + (size_t)t * H;
    float zvi = zi[idx];
    float iv  = zvi * (1.0f / (1.0f + expf(-zvi)));          // silu
    float fv  = lower + upper * (1.0f / (1.0f + expf(-zf[idx])));
    float gv  = 1.0f / (1.0f + expf(-zg[idx]));
    hs = fv * hs + (1.0f - fv) * iv;
    ybf[idx] = (__bf16)(hs * gv);
  }
  hlast[tid] = hs;
}

// ---------------------------------------------------------------------------
// bf16 WMMA GEMM:  C[M,N](f32) = A[M,K](bf16) @ B[K,N] + bias (+ residual)
//   B is supplied transposed: Bt[Npad,K] bf16, Npad = gridDim.y*128.
//   Block: 256 threads = 8 waves; block tile 128x128; wave tile 32x64
//   (2x4 16x16 accum tiles). K stepped by 32 (one v_wmma_f32_16x16x32_bf16
//   per tile per step).
// Template flags make the epilogue branch-free:
//   HAS_RES: fp32 residual add (may alias C)   HAS_CBF: bf16 shadow output
//   NT_OUT : non-temporal stores for C (write-once streaming output; keeps
//            Wp / X resident in L2 for the 50000-wide projection)
// A fragment per ISA 16-bit A 16x32 layout: lane half selects K offset {0,8},
//   elements 0..7 -> K kb..kb+7, elements 8..15 -> K kb+16..kb+23.
// B fragment: same K order as A (lane half selects K offset {0,8}) in each
//   lane (read from Bt row = output column).
// EXEC must be all-ones at WMMA: grid covers padded tiles exactly; bounds
// checks live only in the store epilogue.
// ---------------------------------------------------------------------------
template <bool HAS_RES, bool HAS_CBF, bool NT_OUT>
__global__ __launch_bounds__(256) void gemm_bf16_wmma(
    const __bf16* __restrict__ A,        // [M,K]
    const __bf16* __restrict__ Bt,       // [Npad,K]
    const float*  __restrict__ bias,     // [N]
    const float*  __restrict__ residual, // [M,N] (ignored unless HAS_RES)
    float*        __restrict__ C,        // [M,N]
    __bf16*       __restrict__ Cbf,      // [M,N] (ignored unless HAS_CBF)
    int M, int N, int K) {
  const int lane = threadIdx.x & 31;
  const int wave = threadIdx.x >> 5;
  const int lm   = lane & 15;
  const int hiL  = lane >> 4;                   // lane half: 0 or 1
  const int m0 = blockIdx.x * 128 + (wave >> 1) * 32;
  const int n0 = blockIdx.y * 128 + (wave & 1) * 64;

  const v8f zacc = {0.f, 0.f, 0.f, 0.f, 0.f, 0.f, 0.f, 0.f};
  v8f acc[2][4];
#pragma unroll
  for (int mi = 0; mi < 2; ++mi)
#pragma unroll
    for (int ni = 0; ni < 4; ++ni) acc[mi][ni] = zacc;

  const int koffA = hiL * 8;    // A: lanes 16-31 start at K+8 (interleaved halves)
  const int koffB = hiL * 8;    // B: same interleaved K order as A

  for (int k0 = 0; k0 < K; k0 += 32) {
    v16bf afrag[2], bfrag[4];
#pragma unroll
    for (int mi = 0; mi < 2; ++mi) {
      const __bf16* ap = A + (size_t)(m0 + mi * 16 + lm) * K + (k0 + koffA);
      v8bf lo  = *(const v8bf*)(ap);         // K kb .. kb+7
      v8bf hi8 = *(const v8bf*)(ap + 16);    // K kb+16 .. kb+23
      afrag[mi] = __builtin_shufflevector(lo, hi8,
          0, 1, 2, 3, 4, 5, 6, 7, 8, 9, 10, 11, 12, 13, 14, 15);
    }
#pragma unroll
    for (int ni = 0; ni < 4; ++ni) {
      const __bf16* bp = Bt + (size_t)(n0 + ni * 16 + lm) * K + (k0 + koffB);
      bfrag[ni] = __builtin_shufflevector(*(const v8bf*)(bp), *(const v8bf*)(bp + 16),
          0, 1, 2, 3, 4, 5, 6, 7, 8, 9, 10, 11, 12, 13, 14, 15);
    }
#pragma unroll
    for (int mi = 0; mi < 2; ++mi)
#pragma unroll
      for (int ni = 0; ni < 4; ++ni)
        acc[mi][ni] = __builtin_amdgcn_wmma_f32_16x16x32_bf16(
            false, afrag[mi], false, bfrag[ni], (short)0, acc[mi][ni],
            false, false);
  }

  // Epilogue: D layout — VGPR r holds (M = 16*mi + 8*laneHalf + r, N = lm).
#pragma unroll
  for (int mi = 0; mi < 2; ++mi) {
#pragma unroll
    for (int ni = 0; ni < 4; ++ni) {
      int n = n0 + ni * 16 + lm;
      if (n < N) {
        float bv = bias[n];
#pragma unroll
        for (int r = 0; r < 8; ++r) {
          int m = m0 + mi * 16 + hiL * 8 + r;
          size_t off = (size_t)m * N + n;
          float v = acc[mi][ni][r] + bv;
          if (HAS_RES) v += residual[off];
          if (NT_OUT) {
            __builtin_nontemporal_store(v, &C[off]);
          } else {
            C[off] = v;
          }
          if (HAS_CBF) Cbf[off] = (__bf16)v;
        }
      }
    }
  }
}

// ---------------------------------------------------------------------------
// Host-side orchestration (all launches on `stream`, graph-capture safe).
// ---------------------------------------------------------------------------
extern "C" void kernel_launch(void* const* d_in, const int* in_sizes, int n_in,
                              void* d_out, int out_size, void* d_ws, size_t ws_size,
                              hipStream_t stream) {
  (void)in_sizes; (void)n_in; (void)out_size; (void)ws_size;
  const int*   ids = (const int*)d_in[0];
  const float* emb = (const float*)d_in[1];
  const float* Wi  = (const float*)d_in[2];
  const float* b_i = (const float*)d_in[3];
  const float* Wf  = (const float*)d_in[4];
  const float* b_f = (const float*)d_in[5];
  const float* Wg  = (const float*)d_in[6];
  const float* b_g = (const float*)d_in[7];
  const float* Wo  = (const float*)d_in[8];
  const float* b_o = (const float*)d_in[9];
  const float* Wp  = (const float*)d_in[10];
  const float* b_p = (const float*)d_in[11];
  float* out = (float*)d_out;

  const int H = 256, B = 4, T = 1024, V = 50000, L = 2;
  const int M = B * T;            // 4096
  const int VPAD = 50048;         // 391 * 128

  // Workspace layout (bytes), ~45.5 MB total:
  char* ws = (char*)d_ws;
  float*  xf    = (float*) (ws + 0);          //  4 MB  x fp32 [M,H]
  __bf16* xbf   = (__bf16*)(ws + 4194304);    //  2 MB  x bf16 [M,H]
  float*  zi    = (float*) (ws + 6291456);    //  4 MB  preact i [M,H]
  float*  zf    = (float*) (ws + 10485760);   //  4 MB  preact f [M,H]
  float*  zg    = (float*) (ws + 14680064);   //  4 MB  preact g [M,H]
  __bf16* ybf   = (__bf16*)(ws + 18874368);   //  2 MB  h*g bf16 [M,H]
  __bf16* wg_bf = (__bf16*)(ws + 20971520);   //  1 MB  8x [H,H] bf16 (transposed)
  __bf16* wp_bf = (__bf16*)(ws + 22020096);   // 24.4MB [VPAD,H] bf16 (transposed)

  // 1) Convert + transpose weights to bf16 [N,K] rows.
  for (int l = 0; l < L; ++l) {
    const float* mats[4] = {Wi + (size_t)l * H * H, Wf + (size_t)l * H * H,
                            Wg + (size_t)l * H * H, Wo + (size_t)l * H * H};
    for (int w = 0; w < 4; ++w) {
      convT_kernel<<<(H * H) / 256, 256, 0, stream>>>(
          mats[w], wg_bf + (size_t)(l * 4 + w) * H * H, H, H, H);
    }
  }
  convT_kernel<<<(VPAD * H) / 256, 256, 0, stream>>>(Wp, wp_bf, H, V, VPAD);

  // 2) Embedding gather.
  embed_kernel<<<(M * H) / 256, 256, 0, stream>>>(ids, emb, xf, xbf, M, H);

  // 3) HGRN layers.
  dim3 gate_grid(M / 128, H / 128);   // (32, 2)
  for (int l = 0; l < L; ++l) {
    const __bf16* wtl = wg_bf + (size_t)(l * 4) * H * H;
    gemm_bf16_wmma<false, false, false><<<gate_grid, 256, 0, stream>>>(
        xbf, wtl + 0 * (size_t)H * H, b_i + l * H, nullptr, zi, nullptr, M, H, H);
    gemm_bf16_wmma<false, false, false><<<gate_grid, 256, 0, stream>>>(
        xbf, wtl + 1 * (size_t)H * H, b_f + l * H, nullptr, zf, nullptr, M, H, H);
    gemm_bf16_wmma<false, false, false><<<gate_grid, 256, 0, stream>>>(
        xbf, wtl + 2 * (size_t)H * H, b_g + l * H, nullptr, zg, nullptr, M, H, H);

    float lower = (float)l / (float)L;
    float* hlast = out + (size_t)M * V + (size_t)l * B * H;   // hiddens [L,B,H]
    scan_kernel<<<(B * H) / 256, 256, 0, stream>>>(zi, zf, zg, ybf, hlast,
                                                   B, T, H, lower);

    // out = y @ Wo + bo; x = x + out (fp32 residual, in-place) + bf16 shadow
    gemm_bf16_wmma<true, true, false><<<gate_grid, 256, 0, stream>>>(
        ybf, wtl + 3 * (size_t)H * H, b_o + l * H, xf, xf, xbf, M, H, H);
  }

  // 4) Logits projection: [4096,256] @ [256,50000] -> d_out (fp32, streaming).
  dim3 proj_grid(M / 128, VPAD / 128);   // (32, 391)
  gemm_bf16_wmma<false, false, true><<<proj_grid, 256, 0, stream>>>(
      xbf, wp_bf, b_p, nullptr, out, nullptr, M, V, H);
}
